// EwaldPotential_17162689314996
// MI455X (gfx1250) — matrix operand order repaired and checked by reading, not compiled, not run on hardware
//
#include <hip/hip_runtime.h>
#include <hip/hip_bf16.h>
#include <math.h>

// ---------------------------------------------------------------------------
// Ewald potential attention for MI455X (gfx1250, wave32, WMMA).
// All heavy linear algebra goes through v_wmma_f32_16x16x32_f16, and every
// fragment load is an aligned 16/32-byte vector load (layouts chosen so each
// lane's fragment halves are contiguous in memory).
// ---------------------------------------------------------------------------

typedef __attribute__((ext_vector_type(16))) _Float16 v16h;
typedef __attribute__((ext_vector_type(8)))  _Float16 v8h;
typedef __attribute__((ext_vector_type(8)))  float    v8f;

#define B_    8
#define NPG   2048
#define DIM   64
#define NTOT  (B_ * NPG)
#define KVAL  726            // 6 * 11 * 11 (kx>=0 half-space, radius-5 grid)
#define KP    736            // padded to 46 tiles of 16
#define KT    46             // KP / 16
#define NT    128            // NPG / 16

__constant__ const float TWOPI_F    = 6.28318530717958647692f;
__constant__ const float TWOPI_SQ_F = 39.478417604357434f;
__constant__ const float KSQ_MAX_F  = 2.4674011002723395f;   // (2*pi/4)^2

// ---- CDNA5 WMMA wave32 fragment helpers (per ISA 7.12.2 layouts) ----------
// A (16x32 f16): lane holds row m = lane%16; halves 0-7 -> K = 8*(lane/16)+0..7,
// halves 8-15 -> K = 8*(lane/16)+16..23.  Two aligned 16B loads per lane.
__device__ __forceinline__ v16h load_a_frag(const _Float16* __restrict__ p, int lane) {
    const int off = (lane >> 4) << 3;
    v8h lo = *(const v8h*)(p + off);
    v8h hi = *(const v8h*)(p + off + 16);
    v16h r;
#pragma unroll
    for (int i = 0; i < 8; ++i) { r[i] = lo[i]; r[i + 8] = hi[i]; }
    return r;
}
// B (32x16 f16): lane holds column n = lane%16; halves 0-7 -> K = 8*(lane/16)+0..7,
// halves 8-15 -> K = 8*(lane/16)+16..23.  Two aligned 16B loads per lane (K contiguous).
__device__ __forceinline__ v16h load_b_frag(const _Float16* __restrict__ p, int lane) {
    const int off = (lane >> 4) << 3;
    v8h lo = *(const v8h*)(p + off);
    v8h hi = *(const v8h*)(p + off + 16);
    v16h r;
#pragma unroll
    for (int i = 0; i < 8; ++i) { r[i] = lo[i]; r[i + 8] = hi[i]; }
    return r;
}
// C/D (16x16 f32): vgpr v, lane -> row m = v + 8*(lane/16), col n = lane%16.

__device__ __forceinline__ v8f wmma_f16(v16h a, v16h b, v8f c) {
    return __builtin_amdgcn_wmma_f32_16x16x32_f16(
        /*neg_a=*/false, a, /*neg_b=*/false, b,
        /*c_mod=*/(short)0, c, /*reuse_a=*/false, /*reuse_b=*/false);
}

__device__ __forceinline__ void decode_kvec(int k, float& kx, float& ky, float& kz) {
    int ix = k / 121;
    int t  = k - ix * 121;
    int iy = t / 11 - 5;
    int iz = t - (t / 11) * 11 - 5;
    kx = (float)ix; ky = (float)iy; kz = (float)iz;
}

// ---------------------------------------------------------------------------
// Kernel 1: f32 -> f16.  q stays row-major [n][d] (A-operand).
// k, v are written TRANSPOSED [b][d][n] (B-operand of GEMM1: K-dim = n).
// ---------------------------------------------------------------------------
__global__ void __launch_bounds__(256)
cvt_kernel(const float* __restrict__ q, const float* __restrict__ k,
           const float* __restrict__ v,
           _Float16* __restrict__ qh, _Float16* __restrict__ khT,
           _Float16* __restrict__ vhT) {
    size_t i = (size_t)blockIdx.x * blockDim.x + threadIdx.x;
    if (i >= (size_t)NTOT * DIM) return;
    int d = (int)(i % DIM);
    int n = (int)((i / DIM) % NPG);
    int b = (int)(i / ((size_t)DIM * NPG));
    qh[i] = (_Float16)q[i];
    size_t t = ((size_t)b * DIM + d) * NPG + n;
    khT[t] = (_Float16)k[i];
    vhT[t] = (_Float16)v[i];
}

// ---------------------------------------------------------------------------
// Kernel 2: c,s tables (f16), layout [b][k][n] (n contiguous) for the
// k_pot/v_pot GEMM A-operand (A = e^T).
// ---------------------------------------------------------------------------
__global__ void __launch_bounds__(256)
cs_kernel(const float* __restrict__ positions, const float* __restrict__ cell,
          _Float16* __restrict__ cbuf, _Float16* __restrict__ sbuf) {
    size_t idx = (size_t)blockIdx.x * blockDim.x + threadIdx.x;
    const size_t total = (size_t)B_ * KP * NPG;
    if (idx >= total) return;
    int n = (int)(idx % NPG);
    int k = (int)((idx / NPG) % KP);
    int b = (int)(idx / ((size_t)NPG * KP));
    if (k >= KVAL) { cbuf[idx] = (_Float16)0.0f; sbuf[idx] = (_Float16)0.0f; return; }
    float kx, ky, kz; decode_kvec(k, kx, ky, kz);
    float Lx = cell[b * 9 + 0], Ly = cell[b * 9 + 4], Lz = cell[b * 9 + 8];
    const float* r = positions + ((size_t)b * NPG + n) * 3;
    float ph = TWOPI_F * (r[0] / Lx * kx + r[1] / Ly * ky + r[2] / Lz * kz);
    float sv, cv; __sincosf(ph, &sv, &cv);
    cbuf[idx] = (_Float16)cv;
    sbuf[idx] = (_Float16)sv;
}

// ---------------------------------------------------------------------------
// Kernel 3: k_pot / v_pot GEMMs.  D[16x64] = (c|s)^T[16xN] x (k|v)[Nx64].
// One block per (b, k-tile); 4 waves = {c,s} x {k,v}; 4 d-tiles per wave.
// k_pot results stored [b][k][d] (GEMM2 B-operand, K-dim = d contiguous).
// v_pot results stored [b][d][k] (GEMM3 B-operand, K-dim = k contiguous),
// written as aligned v8h stores straight from the C fragment.
// ---------------------------------------------------------------------------
__global__ void __launch_bounds__(128)
kvpot_gemm_kernel(const _Float16* __restrict__ cbuf, const _Float16* __restrict__ sbuf,
                  const _Float16* __restrict__ khT, const _Float16* __restrict__ vhT,
                  _Float16* __restrict__ kpr, _Float16* __restrict__ kpi,
                  _Float16* __restrict__ vprT, _Float16* __restrict__ vpiT) {
    const int b    = blockIdx.x / KT;
    const int kt   = blockIdx.x % KT;
    const int wave = threadIdx.x >> 5;
    const int lane = threadIdx.x & 31;
    const int srcS = wave & 1;   // 0 -> cos, 1 -> sin
    const int matV = wave >> 1;  // 0 -> k,   1 -> v
    const int m16  = lane & 15;

    const _Float16* Arow = (srcS ? sbuf : cbuf)
                           + ((size_t)b * KP + kt * 16 + m16) * NPG;
    const _Float16* Bb   = (matV ? vhT : khT) + (size_t)b * DIM * NPG;
    const _Float16* Bc0  = Bb + (size_t)( 0 + m16) * NPG;
    const _Float16* Bc1  = Bb + (size_t)(16 + m16) * NPG;
    const _Float16* Bc2  = Bb + (size_t)(32 + m16) * NPG;
    const _Float16* Bc3  = Bb + (size_t)(48 + m16) * NPG;

    v8f acc0 = {}, acc1 = {}, acc2 = {}, acc3 = {};
    for (int nc = 0; nc < NPG; nc += 32) {
        // WGP-scope (high temporal locality) prefetch: data is L2-resident,
        // pull the next chunk into the near caches ahead of the WMMA loads.
        __builtin_prefetch(Arow + nc + 256, 0, 3);
        __builtin_prefetch(Bc0 + nc + 256, 0, 3);
        v16h a  = load_a_frag(Arow + nc, lane);
        v16h b0 = load_b_frag(Bc0 + nc, lane);
        v16h b1 = load_b_frag(Bc1 + nc, lane);
        v16h b2 = load_b_frag(Bc2 + nc, lane);
        v16h b3 = load_b_frag(Bc3 + nc, lane);
        acc0 = wmma_f16(a, b0, acc0);
        acc1 = wmma_f16(a, b1, acc1);
        acc2 = wmma_f16(a, b2, acc2);
        acc3 = wmma_f16(a, b3, acc3);
    }
    if (!matV) {
        // k_pot: [b][k][d], d contiguous
        _Float16* dst = (srcS ? kpi : kpr) + ((size_t)b * KP + kt * 16) * DIM;
#pragma unroll
        for (int v = 0; v < 8; ++v) {
            int row = v + ((lane >> 4) << 3);
            dst[(size_t)row * DIM +  0 + m16] = (_Float16)acc0[v];
            dst[(size_t)row * DIM + 16 + m16] = (_Float16)acc1[v];
            dst[(size_t)row * DIM + 32 + m16] = (_Float16)acc2[v];
            dst[(size_t)row * DIM + 48 + m16] = (_Float16)acc3[v];
        }
    } else {
        // v_pot: [b][d][k], k contiguous -> consecutive C vgprs = consecutive k
        _Float16* dstT = (srcS ? vpiT : vprT) + (size_t)b * DIM * KP;
        const int krow0 = kt * 16 + ((lane >> 4) << 3);
        v8h h0, h1, h2, h3;
#pragma unroll
        for (int v = 0; v < 8; ++v) {
            h0[v] = (_Float16)acc0[v]; h1[v] = (_Float16)acc1[v];
            h2[v] = (_Float16)acc2[v]; h3[v] = (_Float16)acc3[v];
        }
        *(v8h*)(dstT + (size_t)( 0 + m16) * KP + krow0) = h0;
        *(v8h*)(dstT + (size_t)(16 + m16) * KP + krow0) = h1;
        *(v8h*)(dstT + (size_t)(32 + m16) * KP + krow0) = h2;
        *(v8h*)(dstT + (size_t)(48 + m16) * KP + krow0) = h3;
    }
}

// ---------------------------------------------------------------------------
// Kernel 4: logits = c * (q@k_pot_re^T) - s * (q@k_pot_im^T), with validity
// mask, fused around the WMMA C-fragment.  One block per (b, n-tile);
// waves stride over the 46 k-tiles.
// ---------------------------------------------------------------------------
__global__ void __launch_bounds__(128)
logits_gemm_kernel(const _Float16* __restrict__ qf16,
                   const _Float16* __restrict__ kpr, const _Float16* __restrict__ kpi,
                   const float* __restrict__ positions, const float* __restrict__ cell,
                   float* __restrict__ logits) {
    const int b    = blockIdx.x >> 7;
    const int nt   = blockIdx.x & 127;
    const int n0   = nt * 16;
    const int wave = threadIdx.x >> 5;
    const int lane = threadIdx.x & 31;
    const int m16  = lane & 15;

    __shared__ float rsh[16][3];
    if (threadIdx.x < 48) {
        int row = threadIdx.x / 3, comp = threadIdx.x % 3;
        rsh[row][comp] = positions[((size_t)b * NPG + n0 + row) * 3 + comp];
    }
    __syncthreads();

    const float Lx = cell[b * 9 + 0], Ly = cell[b * 9 + 4], Lz = cell[b * 9 + 8];

    // q A-fragments for both 32-wide d chunks (rows n0..n0+15)
    const _Float16* qrow = qf16 + ((size_t)b * NPG + n0 + m16) * DIM;
    v16h aq0 = load_a_frag(qrow, lane);
    v16h aq1 = load_a_frag(qrow + 32, lane);

    for (int kt = wave; kt < KT; kt += 4) {
        const int k0 = kt * 16;
        const _Float16* kr = kpr + ((size_t)b * KP + k0 + m16) * DIM;
        const _Float16* ki = kpi + ((size_t)b * KP + k0 + m16) * DIM;
        v16h br0 = load_b_frag(kr, lane);
        v16h br1 = load_b_frag(kr + 32, lane);
        v16h bi0 = load_b_frag(ki, lane);
        v16h bi1 = load_b_frag(ki + 32, lane);

        v8f accA = {}, accB = {};
        accA = wmma_f16(aq0, br0, accA);
        accA = wmma_f16(aq1, br1, accA);
        accB = wmma_f16(aq0, bi0, accB);
        accB = wmma_f16(aq1, bi1, accB);

        // Fused elementwise: logit = c*A - s*B, masked.
        const int gk = k0 + m16;     // this lane's k column
        float kx, ky, kz; decode_kvec(gk < KVAL ? gk : 0, kx, ky, kz);
        float fx = kx / Lx, fy = ky / Ly, fz = kz / Lz;
        float ksq = TWOPI_SQ_F * (fx * fx + fy * fy + fz * fz);
        bool valid = (gk < KVAL) && (ksq <= KSQ_MAX_F) && (ksq > 0.0f);
#pragma unroll
        for (int v = 0; v < 8; ++v) {
            int mrow = v + ((lane >> 4) << 3);
            float ph = TWOPI_F * (rsh[mrow][0] * fx + rsh[mrow][1] * fy + rsh[mrow][2] * fz);
            float sv, cv; __sincosf(ph, &sv, &cv);
            float lg = valid ? (cv * accA[v] - sv * accB[v]) : -__builtin_inff();
            logits[((size_t)b * NPG + n0 + mrow) * KP + gk] = lg;
        }
    }
}

// ---------------------------------------------------------------------------
// Kernel 5: row softmax over KP (one wave per row, 23 values per lane).
// ---------------------------------------------------------------------------
__global__ void __launch_bounds__(256)
softmax_kernel(float* __restrict__ logits) {
    const int row  = blockIdx.x * 8 + (threadIdx.x >> 5);
    const int lane = threadIdx.x & 31;
    float* p = logits + (size_t)row * KP;
    float vals[23];
    float mx = -__builtin_inff();
#pragma unroll
    for (int j = 0; j < 23; ++j) { vals[j] = p[lane + 32 * j]; mx = fmaxf(mx, vals[j]); }
#pragma unroll
    for (int off = 16; off; off >>= 1) mx = fmaxf(mx, __shfl_xor(mx, off, 32));
    float sum = 0.0f;
#pragma unroll
    for (int j = 0; j < 23; ++j) { vals[j] = __expf(vals[j] - mx); sum += vals[j]; }
#pragma unroll
    for (int off = 16; off; off >>= 1) sum += __shfl_xor(sum, off, 32);
    float inv = 1.0f / sum;
#pragma unroll
    for (int j = 0; j < 23; ++j) p[lane + 32 * j] = vals[j] * inv;
}

// ---------------------------------------------------------------------------
// Kernel 6: ac = attn*cos, as = -attn*sin (f16), layout [b][n][k]
// (reuses the c/s buffers -- their [b][k][n] contents are already consumed).
// ---------------------------------------------------------------------------
__global__ void __launch_bounds__(256)
acas_kernel(const float* __restrict__ attn, const float* __restrict__ positions,
            const float* __restrict__ cell,
            _Float16* __restrict__ acb, _Float16* __restrict__ asb) {
    size_t idx = (size_t)blockIdx.x * blockDim.x + threadIdx.x;
    const size_t total = (size_t)B_ * NPG * KP;
    if (idx >= total) return;
    int gk = (int)(idx % KP);
    int n  = (int)((idx / KP) % NPG);
    int b  = (int)(idx / ((size_t)KP * NPG));
    float a = attn[idx];
    if (gk >= KVAL || a == 0.0f) { acb[idx] = (_Float16)0.0f; asb[idx] = (_Float16)0.0f; return; }
    float kx, ky, kz; decode_kvec(gk, kx, ky, kz);
    float Lx = cell[b * 9 + 0], Ly = cell[b * 9 + 4], Lz = cell[b * 9 + 8];
    const float* r = positions + ((size_t)b * NPG + n) * 3;
    float ph = TWOPI_F * (r[0] / Lx * kx + r[1] / Ly * ky + r[2] / Lz * kz);
    float sv, cv; __sincosf(ph, &sv, &cv);
    acb[idx] = (_Float16)(a * cv);
    asb[idx] = (_Float16)(-a * sv);   // negated so output GEMM is pure add
}

// ---------------------------------------------------------------------------
// Kernel 7: out[n,d] = ac @ v_pot_re + (-as) @ v_pot_im  (f32 accumulate).
// One block per (b, n-tile); wave w owns d-tile w.  B operands come from the
// transposed [b][d][k] v_pot buffers -> two 16B contiguous loads per lane.
// ---------------------------------------------------------------------------
__global__ void __launch_bounds__(128)
out_gemm_kernel(const _Float16* __restrict__ acb, const _Float16* __restrict__ asb,
                const _Float16* __restrict__ vprT, const _Float16* __restrict__ vpiT,
                float* __restrict__ out) {
    const int b    = blockIdx.x >> 7;
    const int nt   = blockIdx.x & 127;
    const int n0   = nt * 16;
    const int dt   = threadIdx.x >> 5;
    const int lane = threadIdx.x & 31;
    const int m16  = lane & 15;

    const _Float16* acrow = acb + ((size_t)b * NPG + n0 + m16) * KP;
    const _Float16* asrow = asb + ((size_t)b * NPG + n0 + m16) * KP;
    const _Float16* vrcol = vprT + ((size_t)b * DIM + dt * 16 + m16) * KP;
    const _Float16* vicol = vpiT + ((size_t)b * DIM + dt * 16 + m16) * KP;

    v8f acc = {};
    for (int kc = 0; kc < KP; kc += 32) {
        __builtin_prefetch(acrow + kc + 256, 0, 3);
        v16h aA = load_a_frag(acrow + kc, lane);
        v16h aS = load_a_frag(asrow + kc, lane);
        v16h bR = load_b_frag(vrcol + kc, lane);
        v16h bI = load_b_frag(vicol + kc, lane);
        acc = wmma_f16(aA, bR, acc);
        acc = wmma_f16(aS, bI, acc);
    }
#pragma unroll
    for (int v = 0; v < 8; ++v) {
        int mrow = v + ((lane >> 4) << 3);
        out[((size_t)b * NPG + n0 + mrow) * DIM + dt * 16 + m16] = acc[v];
    }
}

// ---------------------------------------------------------------------------
// Host launcher
// ---------------------------------------------------------------------------
extern "C" void kernel_launch(void* const* d_in, const int* in_sizes, int n_in,
                              void* d_out, int out_size, void* d_ws, size_t ws_size,
                              hipStream_t stream) {
    (void)in_sizes; (void)n_in; (void)out_size; (void)ws_size;
    const float* q_f32 = (const float*)d_in[0];
    const float* k_f32 = (const float*)d_in[1];
    const float* v_f32 = (const float*)d_in[2];
    const float* pos   = (const float*)d_in[3];
    const float* cell  = (const float*)d_in[4];
    // d_in[5] = batch (contiguous equal-size graphs; unused)

    float* out = (float*)d_out;

    // Workspace carve-up (~106 MB total; fits in the 192 MB global L2)
    char* ws = (char*)d_ws;
    const size_t SZ_QKV = (size_t)NTOT * DIM * 2;        // 2 MB each
    const size_t SZ_CS  = (size_t)B_ * KP * NPG * 2;     // 24 MB each
    const size_t SZ_POT = (size_t)B_ * KP * DIM * 2;     // 0.75 MB each
    _Float16* qh   = (_Float16*)(ws);
    _Float16* khT  = (_Float16*)(ws + SZ_QKV);
    _Float16* vhT  = (_Float16*)(ws + 2 * SZ_QKV);
    _Float16* cbuf = (_Float16*)(ws + 3 * SZ_QKV);               // later reused as ac
    _Float16* sbuf = (_Float16*)(ws + 3 * SZ_QKV + SZ_CS);       // later reused as -as
    _Float16* kpr  = (_Float16*)(ws + 3 * SZ_QKV + 2 * SZ_CS);
    _Float16* kpi  = (_Float16*)(ws + 3 * SZ_QKV + 2 * SZ_CS + SZ_POT);
    _Float16* vprT = (_Float16*)(ws + 3 * SZ_QKV + 2 * SZ_CS + 2 * SZ_POT);
    _Float16* vpiT = (_Float16*)(ws + 3 * SZ_QKV + 2 * SZ_CS + 3 * SZ_POT);
    float*    lgts = (float*)   (ws + 3 * SZ_QKV + 2 * SZ_CS + 4 * SZ_POT); // 48 MB

    // 1. f32 -> f16 q/k/v (k,v transposed to [b][d][n])
    cvt_kernel<<<(NTOT * DIM) / 256, 256, 0, stream>>>(q_f32, k_f32, v_f32, qh, khT, vhT);
    // 2. cos/sin tables [b][k][n]
    cs_kernel<<<(int)(((size_t)B_ * KP * NPG) / 256), 256, 0, stream>>>(pos, cell, cbuf, sbuf);
    // 3. k_pot / v_pot  (WMMA)
    kvpot_gemm_kernel<<<B_ * KT, 128, 0, stream>>>(cbuf, sbuf, khT, vhT, kpr, kpi, vprT, vpiT);
    // 4. logits (WMMA + fused phase/mask)
    logits_gemm_kernel<<<B_ * NT, 128, 0, stream>>>(qh, kpr, kpi, pos, cell, lgts);
    // 5. softmax (in place)
    softmax_kernel<<<NTOT / 8, 256, 0, stream>>>(lgts);
    // 6. attn*cos / -attn*sin operands [b][n][k] (reuse c/s buffers)
    acas_kernel<<<(int)(((size_t)B_ * NPG * KP) / 256), 256, 0, stream>>>(lgts, pos, cell, cbuf, sbuf);
    // 7. output GEMM (WMMA)
    out_gemm_kernel<<<B_ * NT, 128, 0, stream>>>(cbuf, sbuf, vprT, vpiT, out);
}
